// _Fact_tt_Attention_9242769621468
// MI455X (gfx1250) — compile-verified, matrix operand order checked by reading, not run on hardware
//
#include <hip/hip_runtime.h>
#include <cstddef>

// ---------------------------------------------------------------------------
// Types for CDNA5 WMMA (wave32).
// ---------------------------------------------------------------------------
typedef __attribute__((ext_vector_type(16))) __bf16    v16bf;
typedef __attribute__((ext_vector_type(8)))  float     v8f;
typedef __attribute__((ext_vector_type(4)))  unsigned  u32x4;

typedef unsigned short bf16_t;   // raw bf16 storage

union BFrag { v16bf v; u32x4 q[2]; };

__device__ __forceinline__ bf16_t f2bf(float f) {
  union { float f; unsigned u; } x; x.f = f;
  unsigned r = x.u + 0x7FFFu + ((x.u >> 16) & 1u);   // round-to-nearest-even
  return (bf16_t)(r >> 16);
}
__device__ __forceinline__ float bf2f(bf16_t b) {
  union { float f; unsigned u; } x; x.u = ((unsigned)b) << 16; return x.f;
}

// Load a 16-bit A/B WMMA fragment (16x32 of bf16) from a row-major
// [row][k] surface.  ISA layout: lane<16 holds K {0..7,16..23},
// lane>=16 holds K {8..15,24..31} of the same 16 rows/cols.
__device__ __forceinline__ v16bf ldfrag(const bf16_t* base, size_t stride,
                                        int row, int kbase) {
  int half = (threadIdx.x >> 4) & 1;
  const bf16_t* p = base + (size_t)row * stride + kbase + half * 8;
  BFrag f;
  f.q[0] = *(const u32x4*)(p);
  f.q[1] = *(const u32x4*)(p + 16);
  return f.v;
}

__device__ __forceinline__ v8f wmma_bf16(v16bf a, v16bf b, v8f c) {
  return __builtin_amdgcn_wmma_f32_16x16x32_bf16(
      false, a, false, b, (short)0, c, false, false);
}

// ---------------------------------------------------------------------------
// Problem constants
// ---------------------------------------------------------------------------
#define SEQ   2304     // 48*48 tokens
#define DIM   768
#define NH    12
#define HD    64
#define RANK  32
#define HH    48       // spatial H == W

// ---------------------------------------------------------------------------
// K0a: Bq = q_FacTs @ FacTv, Bv = v_FacTs @ FacTv   ([32,768] fp32 each)
// ---------------------------------------------------------------------------
__global__ void k_small_br(const float* __restrict__ FacTv,
                           const float* __restrict__ qF,
                           const float* __restrict__ vF,
                           float* __restrict__ Bq, float* __restrict__ Bv) {
  int idx = blockIdx.x * 256 + threadIdx.x;
  if (idx >= RANK * DIM) return;
  int r = idx / DIM, n = idx % DIM;
  float sq = 0.f, sv = 0.f;
  for (int r2 = 0; r2 < RANK; ++r2) {
    float fv = FacTv[r2 * DIM + n];
    sq += qF[r * RANK + r2] * fv;
    sv += vF[r * RANK + r2] * fv;
  }
  Bq[idx] = sq; Bv[idx] = sv;
}

// ---------------------------------------------------------------------------
// K0b: WeffT[n][k] = bf16( Wqkv[k][n] + low-rank update on q/v slices )
// ---------------------------------------------------------------------------
__global__ void k_build_weffT(const float* __restrict__ Wqkv,
                              const float* __restrict__ FacTu,
                              const float* __restrict__ Bq,
                              const float* __restrict__ Bv,
                              bf16_t* __restrict__ WeffT) {
  int idx = blockIdx.x * 256 + threadIdx.x;           // over 768 * 2304
  if (idx >= DIM * 3 * DIM) return;
  int k = idx / (3 * DIM), n = idx % (3 * DIM);
  float v = Wqkv[idx];
  if (n < DIM) {
    float s = 0.f;
    for (int r = 0; r < RANK; ++r) s += FacTu[k * RANK + r] * Bq[r * DIM + n];
    v += s;                                           // FACT_S == 1.0
  } else if (n >= 2 * DIM) {
    float s = 0.f; int nn = n - 2 * DIM;
    for (int r = 0; r < RANK; ++r) s += FacTu[k * RANK + r] * Bv[r * DIM + nn];
    v += s;
  }
  WeffT[(size_t)n * DIM + k] = f2bf(v);
}

// ---------------------------------------------------------------------------
// K0c: fp32 -> bf16 copy
// ---------------------------------------------------------------------------
__global__ void k_f32_to_bf16(const float* __restrict__ src,
                              bf16_t* __restrict__ dst, int n) {
  int idx = blockIdx.x * 256 + threadIdx.x;
  if (idx < n) dst[idx] = f2bf(src[idx]);
}

// K0d: transpose fp32 [rows][cols] -> bf16 [cols][rows]
__global__ void k_transpose_bf16(const float* __restrict__ src,
                                 bf16_t* __restrict__ dst, int rows, int cols) {
  int idx = blockIdx.x * 256 + threadIdx.x;
  if (idx >= rows * cols) return;
  int r = idx / cols, c = idx % cols;
  dst[(size_t)c * rows + r] = f2bf(src[idx]);
}

// ---------------------------------------------------------------------------
// K1/K4: bf16 WMMA GEMM.  C = A[M,K] @ Bt[N,K]^T + bias.
//   mode 0: qkv epilogue -> scatter into qb/kb (head-major [h][m][64]) and
//           vT (head-major channel-major [h][c][2304]); bf16 out.
//   mode 1: fp32 out[m*N + n] = acc + bias[n]
// Block = 256 threads (8 waves), tile BM=128 x BN=64, BK=32.
// Wave (4x2 grid) computes a 32x32 sub-tile = 2x2 WMMA accumulators.
// ---------------------------------------------------------------------------
#define BM 128
#define BN 64
#define BK 32

__global__ __launch_bounds__(256) void k_gemm_bf16(
    const bf16_t* __restrict__ A, const bf16_t* __restrict__ Bt,
    int N, int K, const float* __restrict__ bias,
    float* __restrict__ outf,
    bf16_t* __restrict__ qb, bf16_t* __restrict__ kb, bf16_t* __restrict__ vT,
    int mode) {
  __shared__ bf16_t sA[BM][BK];
  __shared__ bf16_t sB[BN][BK];

  const int tid  = threadIdx.x;
  const int wave = tid >> 5;
  const int li   = tid & 15;
  const int half = (tid >> 4) & 1;
  const int wm   = wave & 3;          // 4 M-strips of 32
  const int wn   = wave >> 2;         // 2 N-strips of 32
  const int m0   = blockIdx.x * BM;
  const int n0   = blockIdx.y * BN;

  v8f acc00 = {}, acc01 = {}, acc10 = {}, acc11 = {};

  // stage indices (A: 16 elems/thread, B: 8 elems/thread)
  const int ar = tid >> 1, ac = (tid & 1) * 16;
  const int br = tid >> 2, bc = (tid & 3) * 8;

  for (int k0 = 0; k0 < K; k0 += BK) {
    const bf16_t* gA = A  + (size_t)(m0 + ar) * K + k0 + ac;
    const bf16_t* gB = Bt + (size_t)(n0 + br) * K + k0 + bc;
    *(u32x4*)&sA[ar][ac]     = *(const u32x4*)(gA);
    *(u32x4*)&sA[ar][ac + 8] = *(const u32x4*)(gA + 8);
    *(u32x4*)&sB[br][bc]     = *(const u32x4*)(gB);
    if (k0 + BK < K)   // CDNA5 global_prefetch_b8 of next A tile
      __builtin_prefetch(A + (size_t)(m0 + ar) * K + k0 + BK + ac, 0, 0);
    __syncthreads();

    v16bf a0 = ldfrag(&sA[0][0], BK, wm * 32 + li,      0);
    v16bf a1 = ldfrag(&sA[0][0], BK, wm * 32 + 16 + li, 0);
    v16bf b0 = ldfrag(&sB[0][0], BK, wn * 32 + li,      0);
    v16bf b1 = ldfrag(&sB[0][0], BK, wn * 32 + 16 + li, 0);

    acc00 = wmma_bf16(a0, b0, acc00);
    acc01 = wmma_bf16(a0, b1, acc01);
    acc10 = wmma_bf16(a1, b0, acc10);
    acc11 = wmma_bf16(a1, b1, acc11);
    __syncthreads();
  }

  // Epilogue.  C layout: lane holds col = li; VGPR r -> row r (lane<16) / r+8.
  #pragma unroll
  for (int i = 0; i < 2; ++i) {
    #pragma unroll
    for (int j = 0; j < 2; ++j) {
      v8f acc = (i == 0) ? (j == 0 ? acc00 : acc01)
                         : (j == 0 ? acc10 : acc11);
      int n = n0 + wn * 32 + j * 16 + li;
      float bv = bias ? bias[n] : 0.f;
      #pragma unroll
      for (int r = 0; r < 8; ++r) {
        int m = m0 + wm * 32 + i * 16 + half * 8 + r;
        float v = acc[r] + bv;
        if (mode == 1) {
          outf[(size_t)m * N + n] = v;
        } else {
          int which = n / DIM;
          int d = n - which * DIM;
          int head = d >> 6, ch = d & 63;
          bf16_t bb = f2bf(v);
          if (which == 0)       qb[((size_t)head * SEQ + m) * HD + ch] = bb;
          else if (which == 1)  kb[((size_t)head * SEQ + m) * HD + ch] = bb;
          else                  vT[((size_t)head * HD + ch) * SEQ + m] = bb;
        }
      }
    }
  }
}

// ---------------------------------------------------------------------------
// K2: decomposed rel-pos bias tables.
//   rel_h[h, m, kh] = q[h,m,:] . rel_pos_h[hq - kh + 47, :]
//   rel_w[h, m, kw] = q[h,m,:] . rel_pos_w[wq - kw + 47, :]
// grid.x = NH*SEQ blocks; 96 threads: t<48 -> rel_h row, t>=48 -> rel_w row.
// ---------------------------------------------------------------------------
__global__ void k_relpos(const bf16_t* __restrict__ qb,
                         const float* __restrict__ rph,
                         const float* __restrict__ rpw,
                         float* __restrict__ relh, float* __restrict__ relw) {
  int hm = blockIdx.x;
  int head = hm / SEQ, m = hm - head * SEQ;
  int hq = m / HH, wq = m - hq * HH;
  const bf16_t* q = qb + ((size_t)head * SEQ + m) * HD;
  int t = threadIdx.x;
  if (t < HH) {
    const float* tab = rph + (size_t)(hq - t + HH - 1) * HD;
    float s = 0.f;
    for (int c = 0; c < HD; ++c) s += bf2f(q[c]) * tab[c];
    relh[((size_t)head * SEQ + m) * HH + t] = s;
  } else {
    int kw = t - HH;
    const float* tab = rpw + (size_t)(wq - kw + HH - 1) * HD;
    float s = 0.f;
    for (int c = 0; c < HD; ++c) s += bf2f(q[c]) * tab[c];
    relw[((size_t)head * SEQ + m) * HH + kw] = s;
  }
}

// ---------------------------------------------------------------------------
// K3: fused flash attention, one head per blockIdx.y.
// 8 waves/block; each wave owns 16 query rows, streams keys in tiles of 32.
// QK^T and PV both via v_wmma_f32_16x16x32_bf16.  Online softmax in fp32.
// P is transposed C-layout -> A-layout through a per-wave LDS staging tile.
// ---------------------------------------------------------------------------
__global__ __launch_bounds__(256) void k_flash_attn(
    const bf16_t* __restrict__ qb, const bf16_t* __restrict__ kb,
    const bf16_t* __restrict__ vT,
    const float* __restrict__ relh, const float* __restrict__ relw,
    bf16_t* __restrict__ attnout) {
  __shared__ bf16_t pbuf[8][16][32];

  const int head = blockIdx.y;
  const int wave = threadIdx.x >> 5;
  const int li   = threadIdx.x & 15;
  const int half = (threadIdx.x >> 4) & 1;
  const int qbase = blockIdx.x * 128 + wave * 16;

  const bf16_t* qh = qb + (size_t)head * SEQ * HD;
  const bf16_t* kh = kb + (size_t)head * SEQ * HD;
  const bf16_t* vh = vT + (size_t)head * HD * SEQ;
  const float*  rh = relh + (size_t)head * SEQ * HH;
  const float*  rw = relw + (size_t)head * SEQ * HH;

  // Q fragments for channel chunks [0,32) and [32,64)
  v16bf aq0 = ldfrag(qh, HD, qbase + li, 0);
  v16bf aq1 = ldfrag(qh, HD, qbase + li, 32);

  float mrun[8], lrun[8];
  #pragma unroll
  for (int r = 0; r < 8; ++r) { mrun[r] = -1e30f; lrun[r] = 0.f; }
  v8f o0 = {}, o1 = {}, o2 = {}, o3 = {};

  for (int kt = 0; kt < SEQ; kt += 32) {
    // ---- S = (Q*scale) K^T + bias, 16x32 as two 16x16 accumulators ----
    v8f s0 = {}, s1 = {};
    s0 = wmma_bf16(aq0, ldfrag(kh, HD, kt + li,      0),  s0);
    s0 = wmma_bf16(aq1, ldfrag(kh, HD, kt + li,      32), s0);
    s1 = wmma_bf16(aq0, ldfrag(kh, HD, kt + 16 + li, 0),  s1);
    s1 = wmma_bf16(aq1, ldfrag(kh, HD, kt + 16 + li, 32), s1);

    const int key0 = kt + li, key1 = kt + 16 + li;
    const int kh0 = key0 / HH, kw0 = key0 - kh0 * HH;
    const int kh1 = key1 / HH, kw1 = key1 - kh1 * HH;

    float p0[8], p1[8], alpha[8];
    #pragma unroll
    for (int r = 0; r < 8; ++r) {
      int qrow = qbase + half * 8 + r;
      float v0 = s0[r] * 0.125f + rh[(size_t)qrow * HH + kh0]
                                + rw[(size_t)qrow * HH + kw0];
      float v1 = s1[r] * 0.125f + rh[(size_t)qrow * HH + kh1]
                                + rw[(size_t)qrow * HH + kw1];
      float mx = fmaxf(v0, v1);
      mx = fmaxf(mx, __shfl_xor(mx, 1, 32));
      mx = fmaxf(mx, __shfl_xor(mx, 2, 32));
      mx = fmaxf(mx, __shfl_xor(mx, 4, 32));
      mx = fmaxf(mx, __shfl_xor(mx, 8, 32));
      float mnew = fmaxf(mrun[r], mx);
      p0[r] = __expf(v0 - mnew);
      p1[r] = __expf(v1 - mnew);
      float rs = p0[r] + p1[r];
      rs += __shfl_xor(rs, 1, 32);
      rs += __shfl_xor(rs, 2, 32);
      rs += __shfl_xor(rs, 4, 32);
      rs += __shfl_xor(rs, 8, 32);
      alpha[r] = __expf(mrun[r] - mnew);
      lrun[r]  = lrun[r] * alpha[r] + rs;
      mrun[r]  = mnew;
    }
    #pragma unroll
    for (int r = 0; r < 8; ++r) {
      o0[r] *= alpha[r]; o1[r] *= alpha[r];
      o2[r] *= alpha[r]; o3[r] *= alpha[r];
    }

    // ---- P: C-layout -> A-layout via per-wave LDS tile ----
    #pragma unroll
    for (int r = 0; r < 8; ++r) {
      pbuf[wave][half * 8 + r][li]      = f2bf(p0[r]);
      pbuf[wave][half * 8 + r][16 + li] = f2bf(p1[r]);
    }
    __syncthreads();
    v16bf ap = ldfrag(&pbuf[wave][0][0], 32, li, 0);

    // ---- O += P @ V  (K=32 keys, 4 channel chunks of 16) ----
    o0 = wmma_bf16(ap, ldfrag(vh, SEQ, 0 * 16 + li, kt), o0);
    o1 = wmma_bf16(ap, ldfrag(vh, SEQ, 1 * 16 + li, kt), o1);
    o2 = wmma_bf16(ap, ldfrag(vh, SEQ, 2 * 16 + li, kt), o2);
    o3 = wmma_bf16(ap, ldfrag(vh, SEQ, 3 * 16 + li, kt), o3);
    __syncthreads();
  }

  // ---- normalize + write bf16 [m][768] for the projection GEMM ----
  #pragma unroll
  for (int r = 0; r < 8; ++r) {
    int m = qbase + half * 8 + r;
    float inv = 1.f / lrun[r];
    size_t base = (size_t)m * DIM + head * HD + li;
    attnout[base + 0]  = f2bf(o0[r] * inv);
    attnout[base + 16] = f2bf(o1[r] * inv);
    attnout[base + 32] = f2bf(o2[r] * inv);
    attnout[base + 48] = f2bf(o3[r] * inv);
  }
}

// ---------------------------------------------------------------------------
// Host-side launcher
// ---------------------------------------------------------------------------
extern "C" void kernel_launch(void* const* d_in, const int* in_sizes, int n_in,
                              void* d_out, int out_size, void* d_ws, size_t ws_size,
                              hipStream_t stream) {
  (void)in_sizes; (void)n_in; (void)out_size; (void)ws_size;

  const float* x      = (const float*)d_in[0];
  const float* Wqkv   = (const float*)d_in[1];
  const float* bqkv   = (const float*)d_in[2];
  const float* FacTu  = (const float*)d_in[3];
  const float* FacTv  = (const float*)d_in[4];
  const float* qFacTs = (const float*)d_in[5];
  const float* vFacTs = (const float*)d_in[6];
  const float* rph    = (const float*)d_in[7];
  const float* rpw    = (const float*)d_in[8];
  const float* Wproj  = (const float*)d_in[9];
  const float* bproj  = (const float*)d_in[10];
  float* out = (float*)d_out;

  char* w = (char*)d_ws;
  size_t off = 0;
  auto nextf = [&](size_t elems) -> float* {
    float* p = (float*)(w + off); off += ((elems * 4 + 255) / 256) * 256; return p;
  };
  auto nextb = [&](size_t elems) -> bf16_t* {
    bf16_t* p = (bf16_t*)(w + off); off += ((elems * 2 + 255) / 256) * 256; return p;
  };

  float*  Bq     = nextf((size_t)RANK * DIM);
  float*  Bv     = nextf((size_t)RANK * DIM);
  bf16_t* WeffT  = nextb((size_t)3 * DIM * DIM);   // [3*dim][dim]
  bf16_t* Xbf    = nextb((size_t)SEQ * DIM);
  bf16_t* Qb     = nextb((size_t)NH * SEQ * HD);
  bf16_t* Kb     = nextb((size_t)NH * SEQ * HD);
  bf16_t* VT     = nextb((size_t)NH * HD * SEQ);
  float*  RelH   = nextf((size_t)NH * SEQ * HH);
  float*  RelW   = nextf((size_t)NH * SEQ * HH);
  bf16_t* AttnO  = nextb((size_t)SEQ * DIM);
  bf16_t* WprojT = nextb((size_t)DIM * DIM);

  // 0) low-rank folds + bf16 conversions
  k_small_br<<<(RANK * DIM + 255) / 256, 256, 0, stream>>>(FacTv, qFacTs, vFacTs, Bq, Bv);
  k_build_weffT<<<(DIM * 3 * DIM + 255) / 256, 256, 0, stream>>>(Wqkv, FacTu, Bq, Bv, WeffT);
  k_f32_to_bf16<<<(SEQ * DIM + 255) / 256, 256, 0, stream>>>(x, Xbf, SEQ * DIM);
  k_transpose_bf16<<<(DIM * DIM + 255) / 256, 256, 0, stream>>>(Wproj, WprojT, DIM, DIM);

  // 1) qkv = x @ Weff + bqkv  (WMMA), scatter q/k/v^T per head
  k_gemm_bf16<<<dim3(SEQ / BM, (3 * DIM) / BN), 256, 0, stream>>>(
      Xbf, WeffT, 3 * DIM, DIM, bqkv, nullptr, Qb, Kb, VT, 0);

  // 2) decomposed rel-pos bias tables
  k_relpos<<<dim3(NH * SEQ), 96, 0, stream>>>(Qb, rph, rpw, RelH, RelW);

  // 3) flash attention (WMMA QK^T and PV, online softmax)
  k_flash_attn<<<dim3(SEQ / 128, NH), 256, 0, stream>>>(Qb, Kb, VT, RelH, RelW, AttnO);

  // 4) out = attn_out @ Wproj + bproj  (WMMA), fp32 result
  k_gemm_bf16<<<dim3(SEQ / BM, DIM / BN), 256, 0, stream>>>(
      AttnO, WprojT, DIM, DIM, bproj, out, nullptr, nullptr, nullptr, 1);
}
